// Attention_84679575208344
// MI455X (gfx1250) — matrix operand order repaired and checked by reading, not compiled, not run on hardware
//
#include <hip/hip_runtime.h>
#include <hip/hip_bf16.h>

typedef __attribute__((ext_vector_type(16))) _Float16 v16h;
typedef __attribute__((ext_vector_type(8)))  float    v8f;

#define NB      8
#define NSEQ    4096
#define CCH     384
#define NHEAD   8
#define HD      48
#define KEPS    1e-3f
#define DEN_EPS 1e-8f
#define DATA_NORM 0.37991784282579627f   // 48^(-0.25)

union FragU { v16h h; unsigned int u[8]; };
union H8   { _Float16 h[8]; uint4 u; };

// A fragment: 16x32 f16 tile, row-major with pitch (halves). ISA layout:
// lane(0-15)=row M, VGPR v<4 -> K=2v+8*hi, v>=4 -> K=16+2(v-4)+8*hi (hi=lane>=16)
__device__ inline v16h load_frag_a(const _Float16* p, int pitch, int lane) {
  FragU f;
  int row = lane & 15;
  int hi  = lane >> 4;
  const unsigned int* q0 = reinterpret_cast<const unsigned int*>(p + row * pitch + hi * 8);
  const unsigned int* q1 = reinterpret_cast<const unsigned int*>(p + row * pitch + 16 + hi * 8);
#pragma unroll
  for (int v = 0; v < 4; ++v) { f.u[v] = q0[v]; f.u[v + 4] = q1[v]; }
  return f.h;
}

// B fragment: 32x16 f16 tile stored as [col][K] with pitch (halves).
// lane(0-15)=col N; VGPR v<4 -> K=2v+8*hi, v>=4 -> K=16+2(v-4)+8*hi (same as A).
__device__ inline v16h load_frag_b(const _Float16* p, int pitch, int lane) {
  FragU f;
  int col = lane & 15;
  int hi  = lane >> 4;
  const unsigned int* q0 = reinterpret_cast<const unsigned int*>(p + col * pitch + hi * 8);
  const unsigned int* q1 = reinterpret_cast<const unsigned int*>(p + col * pitch + 16 + hi * 8);
#pragma unroll
  for (int v = 0; v < 4; ++v) { f.u[v] = q0[v]; f.u[v + 4] = q1[v]; }
  return f.h;
}

__device__ inline v8f wmma_f16(v16h a, v16h b, v8f c) {
  return __builtin_amdgcn_wmma_f32_16x16x32_f16(false, a, false, b, (short)0, c, false, false);
}

// Convert 8 f32 (two float4) -> 8 f16, one b128 LDS store.
__device__ inline void store8(_Float16* dst, float4 a, float4 b) {
  H8 p;
  p.h[0] = (_Float16)a.x; p.h[1] = (_Float16)a.y;
  p.h[2] = (_Float16)a.z; p.h[3] = (_Float16)a.w;
  p.h[4] = (_Float16)b.x; p.h[5] = (_Float16)b.y;
  p.h[6] = (_Float16)b.z; p.h[7] = (_Float16)b.w;
  *reinterpret_cast<uint4*>(dst) = p.u;
}
__device__ inline void cvt_tile8(_Float16* dst, const float* gsrc) {
  float4 a = *reinterpret_cast<const float4*>(gsrc);
  float4 b = *reinterpret_cast<const float4*>(gsrc + 4);
  store8(dst, a, b);
}

// CDNA5 async global->LDS copy (ASYNCcnt path), 16 bytes per lane.
__device__ inline void async_copy_b128(_Float16* lds_dst, const _Float16* gsrc) {
  unsigned int l = (unsigned int)(unsigned long long)lds_dst;   // low 32b = LDS offset
  unsigned long long g = (unsigned long long)gsrc;
  asm volatile("global_load_async_to_lds_b128 %0, %1, off"
               :: "v"(l), "v"(g) : "memory");
}
__device__ inline void async_wait0() {
  asm volatile("s_wait_asynccnt 0x0" ::: "memory");
}

// -------- Stage 1: qkv = x @ Wqkv^T, fused Performer feature-map epilogue ----
// grid (512, 18), block 256 (8 waves). Block tile 64(M) x 64(N), K step 32.
// LDS double-buffered; next tile's global loads issue before current WMMAs.
__global__ __launch_bounds__(256) void qkv_gemm_kernel(
    const float* __restrict__ x, const float* __restrict__ Wqkv,
    _Float16* __restrict__ qp,   // [bh][n][m]   (m = feature, row-major)
    _Float16* __restrict__ kpT,  // [bh][m][n]   (transposed)
    _Float16* __restrict__ vT)   // [bh][d][n]   (transposed)
{
  __shared__ __align__(16) _Float16 As[2][64][32];
  __shared__ __align__(16) _Float16 Bs[2][64][32];
  int t = threadIdx.x, lane = t & 31, w = t >> 5;
  int rs = w & 3, cg = w >> 2;             // row strip 0..3, col group 0..1
  int m0 = blockIdx.x * 64, n0 = blockIdx.y * 64;
  int lrow = t >> 2, lck = (t & 3) * 8;    // tile-fill assignment (8 halves/thread)
  const float* xg = &x[(size_t)(m0 + lrow) * CCH + lck];
  const float* wg = &Wqkv[(size_t)(n0 + lrow) * CCH + lck];

  cvt_tile8(&As[0][lrow][lck], xg);
  cvt_tile8(&Bs[0][lrow][lck], wg);
  __syncthreads();

  v8f acc0 = {}, acc1 = {};
  const int NK = CCH / 32;                 // 12 k-steps
  for (int k = 0; k < NK; ++k) {
    int cur = k & 1, nxt = cur ^ 1;
    bool have = (k + 1 < NK);
    float4 pa0, pa1, pb0, pb1;
    if (have) {                            // prefetch next tile into registers
      const float* xs = xg + (k + 1) * 32;
      const float* wsrc = wg + (k + 1) * 32;
      pa0 = *reinterpret_cast<const float4*>(xs);
      pa1 = *reinterpret_cast<const float4*>(xs + 4);
      pb0 = *reinterpret_cast<const float4*>(wsrc);
      pb1 = *reinterpret_cast<const float4*>(wsrc + 4);
    }
    v16h a  = load_frag_a(&As[cur][rs * 16][0], 32, lane);
    v16h b0 = load_frag_b(&Bs[cur][cg * 32][0], 32, lane);
    v16h b1 = load_frag_b(&Bs[cur][cg * 32 + 16][0], 32, lane);
    acc0 = wmma_f16(a, b0, acc0);
    acc1 = wmma_f16(a, b1, acc1);
    if (have) {                            // drain prefetch into other buffer
      store8(&As[nxt][lrow][lck], pa0, pa1);
      store8(&Bs[nxt][lrow][lck], pb0, pb1);
    }
    __syncthreads();
  }

  int hi = lane >> 4, ln = lane & 15;
#pragma unroll
  for (int v = 0; v < 8; ++v) {
    int rl = rs * 16 + v + 8 * hi;
    int m = m0 + rl;
    int b = m >> 12, n = m & 4095;
#pragma unroll
    for (int f = 0; f < 2; ++f) {
      float val = f ? acc1[v] : acc0[v];
      int d   = n0 + cg * 32 + f * 16 + ln;    // 0..1151
      int s   = d / CCH;
      int rem = d - s * CCH;
      int h   = rem / HD;
      int hd  = rem - h * HD;
      int bh  = b * NHEAD + h;
      if (s == 0) {
        float qv = fmaxf(DATA_NORM * val, 0.f) + KEPS;
        qp[((size_t)bh * NSEQ + n) * HD + hd] = (_Float16)qv;
      } else if (s == 1) {
        float kv = fmaxf(DATA_NORM * val, 0.f) + KEPS;
        kpT[((size_t)bh * HD + hd) * NSEQ + n] = (_Float16)kv;
      } else {
        vT[((size_t)bh * HD + hd) * NSEQ + n] = (_Float16)val;
      }
    }
  }
}

// -------- Stage 2: per (b,h): kptv[d][m] = sum_n vT[d][n]*kpT[m][n]; ks[m] ---
// grid 64, block 256 (8 waves). 3x3 WMMA tiles, K=4096 split across waves,
// partials combined via LDS float atomics (ds_add_f32).
__global__ __launch_bounds__(256) void kptv_ks_kernel(
    const _Float16* __restrict__ kpT, const _Float16* __restrict__ vT,
    _Float16* __restrict__ kptvh,   // [bh][d][m] f16
    float* __restrict__ ksg)        // [bh][m]
{
  __shared__ float red[HD * HD];
  int t = threadIdx.x, lane = t & 31, w = t >> 5;
  int bh = blockIdx.x;
  const _Float16* vp  = vT  + (size_t)bh * HD * NSEQ;
  const _Float16* kpp = kpT + (size_t)bh * HD * NSEQ;
  for (int idx = t; idx < HD * HD; idx += 256) red[idx] = 0.f;
  __syncthreads();
  v8f zero = {};
  v8f acc[3][3];
#pragma unroll
  for (int r = 0; r < 3; ++r)
#pragma unroll
    for (int c = 0; c < 3; ++c) acc[r][c] = zero;
  for (int kk = w * 32; kk < NSEQ; kk += 8 * 32) {
    v16h af[3], bf[3];
#pragma unroll
    for (int r = 0; r < 3; ++r) af[r] = load_frag_a(vp  + (size_t)(r * 16) * NSEQ + kk, NSEQ, lane);
#pragma unroll
    for (int c = 0; c < 3; ++c) bf[c] = load_frag_b(kpp + (size_t)(c * 16) * NSEQ + kk, NSEQ, lane);
#pragma unroll
    for (int r = 0; r < 3; ++r)
#pragma unroll
      for (int c = 0; c < 3; ++c) acc[r][c] = wmma_f16(af[r], bf[c], acc[r][c]);
  }
  int hi = lane >> 4, ln = lane & 15;
#pragma unroll
  for (int r = 0; r < 3; ++r)
#pragma unroll
    for (int c = 0; c < 3; ++c)
#pragma unroll
      for (int v = 0; v < 8; ++v) {
        int row = r * 16 + v + 8 * hi, col = c * 16 + ln;
        atomicAdd(&red[row * HD + col], acc[r][c][v]);
      }
  __syncthreads();
  for (int idx = t; idx < HD * HD; idx += 256)
    kptvh[(size_t)bh * HD * HD + idx] = (_Float16)red[idx];
  if (t < HD) {
    float s = 0.f;
    const _Float16* kr = kpp + (size_t)t * NSEQ;
    for (int n = 0; n < NSEQ; n += 8) {     // vectorized column sum
      H8 p; p.u = *reinterpret_cast<const uint4*>(kr + n);
#pragma unroll
      for (int j = 0; j < 8; ++j) s += (float)p.h[j];
    }
    ksg[bh * HD + t] = s;
  }
}

// -------- Stage 3: out[n][d] = (qp[n]·kptv[d]) / (qp[n]·ks + eps) ------------
// grid (64 row-tiles, 64 bh), block 128 (4 waves). K=48 zero-padded to 64.
// Pure-f16 tiles staged with CDNA5 async global->LDS copies.
__global__ __launch_bounds__(128) void attn_out_kernel(
    const _Float16* __restrict__ qp, const _Float16* __restrict__ kptvh,
    const float* __restrict__ ksg,
    _Float16* __restrict__ y)       // flat (B,H,N,HD) -> rows of final GEMM
{
  __shared__ __align__(16) _Float16 Qs[64][64];   // qp tile, m padded 48->64 with zeros
  __shared__ __align__(16) _Float16 Ks[48][64];   // kptv as [col=d][K=m], padded
  __shared__ float ksv[HD];
  __shared__ float Dv[64];
  int t = threadIdx.x, lane = t & 31, w = t >> 5;
  int bh = blockIdx.y;
  int n0 = blockIdx.x * 64;
  const _Float16* qpb = qp + (size_t)bh * NSEQ * HD;
  const _Float16* kb  = kptvh + (size_t)bh * HD * HD;
  // Qs: 64 rows x 48 halves = 384 x 16B chunks (3 per thread), async to LDS.
#pragma unroll
  for (int i = 0; i < 3; ++i) {
    int cidx = t + i * 128;
    int row = cidx / 6, ch = cidx % 6;
    async_copy_b128(&Qs[row][ch * 8], &qpb[(size_t)(n0 + row) * HD + ch * 8]);
  }
  // Ks: 48x48 halves = 288 x 16B chunks, async to LDS.
  for (int cidx = t; cidx < 288; cidx += 128) {
    int row = cidx / 6, ch = cidx % 6;
    async_copy_b128(&Ks[row][ch * 8], &kb[(size_t)row * HD + ch * 8]);
  }
  // zero padding (m = 48..63) via normal DS stores
  for (int idx = t; idx < 64 * 16; idx += 128) Qs[idx / 16][48 + (idx & 15)] = (_Float16)0.f;
  for (int idx = t; idx < HD * 16; idx += 128) Ks[idx / 16][48 + (idx & 15)] = (_Float16)0.f;
  if (t < HD) ksv[t] = ksg[bh * HD + t];
  async_wait0();
  __syncthreads();
  if (t < 64) {
    float s = 0.f;
#pragma unroll
    for (int m = 0; m < HD; ++m) s += (float)Qs[t][m] * ksv[m];
    Dv[t] = s;
  }
  __syncthreads();
  v8f zero = {};
  v8f acc[3] = {zero, zero, zero};
#pragma unroll
  for (int kk = 0; kk < 64; kk += 32) {
    v16h a = load_frag_a(&Qs[w * 16][kk], 64, lane);
#pragma unroll
    for (int c = 0; c < 3; ++c) {
      v16h b = load_frag_b(&Ks[c * 16][kk], 64, lane);
      acc[c] = wmma_f16(a, b, acc[c]);
    }
  }
  int hi = lane >> 4, ln = lane & 15;
#pragma unroll
  for (int c = 0; c < 3; ++c)
#pragma unroll
    for (int v = 0; v < 8; ++v) {
      int rl = w * 16 + v + 8 * hi;
      int col = c * 16 + ln;
      float val = acc[c][v] / (Dv[rl] + DEN_EPS);
      y[((size_t)bh * NSEQ + n0 + rl) * HD + col] = (_Float16)val;
    }
}

// -------- Stage 4: out = y @ Wproj^T + bproj (fp32 out) ----------------------
// grid (512, 6), block 256 (8 waves). Double-buffered; A tile prefetched via
// CDNA5 async global->LDS copies, B tile register-staged f32->f16.
__global__ __launch_bounds__(256) void proj_gemm_kernel(
    const _Float16* __restrict__ y, const float* __restrict__ Wproj,
    const float* __restrict__ bproj, float* __restrict__ out)
{
  __shared__ __align__(16) _Float16 As[2][64][32];
  __shared__ __align__(16) _Float16 Bs[2][64][32];
  int t = threadIdx.x, lane = t & 31, w = t >> 5;
  int rs = w & 3, cg = w >> 2;
  int m0 = blockIdx.x * 64, n0 = blockIdx.y * 64;
  int lrow = t >> 2, lck = (t & 3) * 8;
  const _Float16* yg = &y[(size_t)(m0 + lrow) * CCH + lck];
  const float* wg = &Wproj[(size_t)(n0 + lrow) * CCH + lck];

  async_copy_b128(&As[0][lrow][lck], yg);        // tile 0, one 16B chunk/thread
  cvt_tile8(&Bs[0][lrow][lck], wg);
  async_wait0();
  __syncthreads();

  v8f acc0 = {}, acc1 = {};
  const int NK = CCH / 32;
  for (int k = 0; k < NK; ++k) {
    int cur = k & 1, nxt = cur ^ 1;
    bool have = (k + 1 < NK);
    float4 pb0, pb1;
    if (have) {                                  // prefetch next tile
      async_copy_b128(&As[nxt][lrow][lck], yg + (k + 1) * 32);
      const float* wsrc = wg + (k + 1) * 32;
      pb0 = *reinterpret_cast<const float4*>(wsrc);
      pb1 = *reinterpret_cast<const float4*>(wsrc + 4);
    }
    v16h a  = load_frag_a(&As[cur][rs * 16][0], 32, lane);
    v16h b0 = load_frag_b(&Bs[cur][cg * 32][0], 32, lane);
    v16h b1 = load_frag_b(&Bs[cur][cg * 32 + 16][0], 32, lane);
    acc0 = wmma_f16(a, b0, acc0);
    acc1 = wmma_f16(a, b1, acc1);
    if (have) store8(&Bs[nxt][lrow][lck], pb0, pb1);
    async_wait0();
    __syncthreads();
  }

  int hi = lane >> 4, ln = lane & 15;
#pragma unroll
  for (int v = 0; v < 8; ++v) {
    int m = m0 + rs * 16 + v + 8 * hi;
#pragma unroll
    for (int f = 0; f < 2; ++f) {
      int col = n0 + cg * 32 + f * 16 + ln;
      out[(size_t)m * CCH + col] = (f ? acc1[v] : acc0[v]) + bproj[col];
    }
  }
}

extern "C" void kernel_launch(void* const* d_in, const int* in_sizes, int n_in,
                              void* d_out, int out_size, void* d_ws, size_t ws_size,
                              hipStream_t stream) {
  (void)in_sizes; (void)n_in; (void)out_size; (void)ws_size;
  const float* x     = (const float*)d_in[0];
  const float* Wqkv  = (const float*)d_in[1];
  const float* Wproj = (const float*)d_in[2];
  const float* bproj = (const float*)d_in[3];
  float* out = (float*)d_out;

  // Workspace carve-up (~101 MB total)
  char* ws = (char*)d_ws;
  size_t off = 0;
  auto alloc = [&](size_t bytes) -> char* {
    char* p = ws + off;
    off = (off + bytes + 255) & ~(size_t)255;
    return p;
  };
  const size_t bhnd = (size_t)NB * NHEAD * NSEQ * HD;     // 12,582,912 elems
  _Float16* qp    = (_Float16*)alloc(bhnd * 2);
  _Float16* kpT   = (_Float16*)alloc(bhnd * 2);
  _Float16* vT    = (_Float16*)alloc(bhnd * 2);
  _Float16* y     = (_Float16*)alloc(bhnd * 2);
  _Float16* kptvh = (_Float16*)alloc((size_t)NB * NHEAD * HD * HD * 2);
  float*    ksg   = (float*)   alloc((size_t)NB * NHEAD * HD * 4);

  hipLaunchKernelGGL(qkv_gemm_kernel, dim3(512, 18), dim3(256), 0, stream,
                     x, Wqkv, qp, kpT, vT);
  hipLaunchKernelGGL(kptv_ks_kernel, dim3(64), dim3(256), 0, stream,
                     kpT, vT, kptvh, ksg);
  hipLaunchKernelGGL(attn_out_kernel, dim3(64, 64), dim3(128), 0, stream,
                     qp, kptvh, ksg, y);
  hipLaunchKernelGGL(proj_gemm_kernel, dim3(512, 6), dim3(256), 0, stream,
                     y, Wproj, bproj, out);
}
